// LSTM_39058432589846
// MI455X (gfx1250) — compile-verified, matrix operand order checked by reading, not run on hardware
//
#include <hip/hip_runtime.h>

// ---------------------------------------------------------------------------
// 2-layer LSTM, B=64 T=512 D=H=1024.
//   Phase 0: pack x -> bf16 fragment-native slabs; pack 4 weights likewise.
//   Phase 1: Gin0 = x @ w_ih0^T          (big WMMA GEMM, bf16->f32 acc)
//   Phase 2: per t: gates = Gin0[t] + h0 @ w_hh0^T -> pointwise cell update
//   Phase 3: Gin1 = h0_all @ w_ih1^T     (big WMMA GEMM)
//   Phase 4: per t: gates = Gin1[t] + h1 @ w_hh1^T -> pointwise (writes out)
//
// Both GEMM operands are pre-packed into the exact per-lane VGPR layout of
// v_wmma_f32_16x16x32_bf16 (CDNA5 ISA 7.12.2): per (16 x 32) tile each lane's
// 16 bf16 are contiguous (32B). The GEMM is LDS-free and barrier-free:
// global_load_b128 + WMMA only. A sched_barrier between the next chunk's
// prefetch loads and the current chunk's WMMAs pins the software pipeline so
// waits stay partial (loads in flight across WMMA groups).
// ---------------------------------------------------------------------------

typedef __attribute__((ext_vector_type(16))) __bf16 v16bf;
typedef __attribute__((ext_vector_type(8)))  float  v8f;

// global-memory (addrspace 1) fragment pointer: forces global_load_b128
typedef const v16bf __attribute__((address_space(1)))* gfrag_p;

static constexpr int LB = 64;     // batch
static constexpr int LT = 512;    // seq len
static constexpr int LD = 1024;   // input size
static constexpr int LH = 1024;   // hidden size
static constexpr int LG = 4096;   // 4*H gates
static constexpr int KTC = 32;    // K chunks of 32 (K = 1024 everywhere)

__device__ inline v16bf load_frag(const __bf16* p) {
    return *(gfrag_p)(const v16bf*)p;
}

// ---------------------------------------------------------------- utilities
__global__ void zero_u32_kernel(unsigned* __restrict__ p, int n) {
    int i = blockIdx.x * blockDim.x + threadIdx.x;
    if (i < n) p[i] = 0u;
}

// Fragment element index e (0..15) + lane-group grp (0/1) -> K offset within
// a 32-deep chunk, per CDNA5 ISA 7.12.2 16-bit operand layout.
__host__ __device__ inline int frag_k_rel(int e, int grp) {
    int v = e >> 1, half = e & 1;
    return grp * 8 + ((v < 4) ? (2 * v + half) : (16 + 2 * (v - 4) + half));
}

// w: (4096 n, 1024 k) row-major fp32 -> packed bf16:
//   wp[((ntile*KTC + kt)*32 + lane)*16 + e], n = ntile*16+(l&15),
//   k = kt*32 + frag_k_rel(e, l>>4)
__global__ void pack_weight_kernel(const float* __restrict__ w,
                                   __bf16* __restrict__ wp) {
    int idx = blockIdx.x * blockDim.x + threadIdx.x;   // 4096*1024
    if (idx >= LG * LH) return;
    int e     = idx & 15;
    int l     = (idx >> 4) & 31;
    int kt    = (idx >> 9) & 31;
    int ntile = idx >> 14;
    int n = ntile * 16 + (l & 15);
    int k = kt * 32 + frag_k_rel(e, l >> 4);
    wp[idx] = (__bf16)w[(size_t)n * LH + k];
}

// x: (B,T,D) fp32 -> packed bf16 A-slabs, one 64-row slab per t:
//   xp[s*65536 + mt*16384 + kt*512 + l*16 + e], row b = mt*16+(l&15),
//   k = kt*32 + frag_k_rel(e, l>>4), slab s = t.
__global__ void pack_x_kernel(const float* __restrict__ x,
                              __bf16* __restrict__ xp) {
    int idx = blockIdx.x * blockDim.x + threadIdx.x;   // B*T*D = 2^25
    if (idx >= LB * LT * LD) return;
    int e  = idx & 15;
    int l  = (idx >> 4) & 31;
    int kt = (idx >> 9) & 31;
    int mt = (idx >> 14) & 3;
    int s  = idx >> 16;                               // t
    int b = mt * 16 + (l & 15);
    int k = kt * 32 + frag_k_rel(e, l >> 4);
    xp[idx] = (__bf16)x[((size_t)b * LT + s) * LD + k];
}

// ------------------------------------------------------------- WMMA GEMM
// C[s*64+row][col] = sum_k A[row][k] * W[k][col] (+ gin), both packed.
// LDS-free / barrier-free; software-pipelined fragment loads pinned by
// sched_barrier.
template <int NT, int WAVES>
__global__ __launch_bounds__(WAVES * 32)
void gemm_wmma_packed(const __bf16* __restrict__ Ap,  // packed A slabs
                      const __bf16* __restrict__ Bp,  // packed weights
                      __bf16* __restrict__ out_bf,    // bf16 output (or null)
                      float*  __restrict__ out_f,     // f32 output (or null)
                      const __bf16* __restrict__ gin) // bf16 addend (or null)
{
    constexpr int MSUB = 4;                 // 64-row M tile
    constexpr int WVN  = WAVES / MSUB;      // waves sharing one m-subtile
    constexpr int ACC  = (NT / 16) / WVN;   // n-subtiles per wave

    const int tid   = threadIdx.x;
    const int w     = tid >> 5;
    const int lane  = tid & 31;
    const int m_idx = w % MSUB;
    const int n_grp = w / MSUB;

    const int tile_n0 = blockIdx.x * NT;
    const int s       = blockIdx.y;         // M-slab index

    const __bf16* aptr =
        Ap + (((size_t)(s * MSUB + m_idx) * KTC) * 32 + lane) * 16;
    const __bf16* bptr[ACC];
#pragma unroll
    for (int j = 0; j < ACC; ++j) {
        int ntile = (tile_n0 >> 4) + n_grp * ACC + j;
        bptr[j] = Bp + (((size_t)ntile * KTC) * 32 + lane) * 16;
    }

    v8f acc[ACC];
#pragma unroll
    for (int j = 0; j < ACC; ++j)
#pragma unroll
        for (int e = 0; e < 8; ++e) acc[j][e] = 0.0f;

    // ---- software-pipelined main loop (prefetch distance 1)
    v16bf afc = load_frag(aptr);
    v16bf bfc[ACC];
#pragma unroll
    for (int j = 0; j < ACC; ++j) bfc[j] = load_frag(bptr[j]);

#pragma unroll 4
    for (int kt = 0; kt < KTC - 1; ++kt) {
        const int koff = (kt + 1) * 512;     // elems per 32-lane chunk
        v16bf afn = load_frag(aptr + koff);
        v16bf bfn[ACC];
#pragma unroll
        for (int j = 0; j < ACC; ++j) bfn[j] = load_frag(bptr[j] + koff);
        // Pin the pipeline: prefetch loads may not sink below this point,
        // WMMAs may not hoist above it.
        __builtin_amdgcn_sched_barrier(0);
#pragma unroll
        for (int j = 0; j < ACC; ++j)
            acc[j] = __builtin_amdgcn_wmma_f32_16x16x32_bf16(
                false, afc, false, bfc[j], (short)0, acc[j], false, false);
        __builtin_amdgcn_sched_barrier(0);
        afc = afn;
#pragma unroll
        for (int j = 0; j < ACC; ++j) bfc[j] = bfn[j];
    }
#pragma unroll
    for (int j = 0; j < ACC; ++j)
        acc[j] = __builtin_amdgcn_wmma_f32_16x16x32_bf16(
            false, afc, false, bfc[j], (short)0, acc[j], false, false);

    // ---- epilogue (ISA 32-bit C/D 16x16 layout)
    {
        const int grp  = lane >> 4;
        const int coll = lane & 15;
#pragma unroll
        for (int j = 0; j < ACC; ++j) {
            const int ns = (n_grp * ACC + j) * 16;
#pragma unroll
            for (int r = 0; r < 8; ++r) {
                const int row = s * 64 + m_idx * 16 + grp * 8 + r;
                const int col = tile_n0 + ns + coll;
                float vv = acc[j][r];
                if (gin) vv += (float)gin[(size_t)row * LG + col];
                if (out_f) out_f[(size_t)row * LG + col] = vv;
                else       out_bf[(size_t)row * LG + col] = (__bf16)vv;
            }
        }
    }
}

// ------------------------------------------------------ LSTM cell pointwise
// Writes new h both as fp32 (outputs/finals) and as packed-A bf16 slab for
// the next step's WMMA GEMM.
__global__ void lstm_pointwise_kernel(const float* __restrict__ gates,  // B x 4H
                                      float* __restrict__ c,            // B x H
                                      __bf16* __restrict__ h_packed,    // 64x1024 slab
                                      float* __restrict__ out_bth,      // or null
                                      int t,
                                      float* __restrict__ h_final,      // or null
                                      float* __restrict__ c_final)      // or null
{
    int idx = blockIdx.x * blockDim.x + threadIdx.x;
    if (idx >= LB * LH) return;
    int b = idx / LH;
    int h = idx % LH;
    const float* g = gates + (size_t)b * LG;
    float gi = g[h];
    float gf = g[LH + h];
    float gg = g[2 * LH + h];
    float go = g[3 * LH + h];
    float si = 1.0f / (1.0f + expf(-gi));
    float sf = 1.0f / (1.0f + expf(-gf));
    float so = 1.0f / (1.0f + expf(-go));
    float c_new = sf * c[idx] + si * tanhf(gg);
    float h_new = so * tanhf(c_new);
    c[idx] = c_new;

    // packed-A offset for (row b, k = h)
    int mt    = b >> 4;
    int kt    = h >> 5;
    int k_rel = h & 31;
    int grp   = (k_rel >> 3) & 1;
    int e     = ((k_rel >> 4) << 3) | (k_rel & 7);
    int l     = grp * 16 + (b & 15);
    h_packed[mt * 16384 + kt * 512 + l * 16 + e] = (__bf16)h_new;

    if (out_bth) out_bth[(size_t)b * LT * LH + (size_t)t * LH + h] = h_new;
    if (h_final) h_final[idx] = h_new;
    if (c_final) c_final[idx] = c_new;
}

// ---------------------------------------------------------------- launcher
extern "C" void kernel_launch(void* const* d_in, const int* in_sizes, int n_in,
                              void* d_out, int out_size, void* d_ws, size_t ws_size,
                              hipStream_t stream) {
    const float* x      = (const float*)d_in[0];
    const float* w_ih_0 = (const float*)d_in[1];
    const float* w_hh_0 = (const float*)d_in[2];
    const float* w_ih_1 = (const float*)d_in[3];
    const float* w_hh_1 = (const float*)d_in[4];
    float* out = (float*)d_out;

    // d_out layout: out (B,T,H) | h0_final | h1_final | c0_final | c1_final
    const size_t OUT_BTH = 0;
    const size_t OFF_H0  = (size_t)LB * LT * LH;
    const size_t OFF_H1  = OFF_H0 + (size_t)LB * LH;
    const size_t OFF_C0  = OFF_H1 + (size_t)LB * LH;
    const size_t OFF_C1  = OFF_C0 + (size_t)LB * LH;

    // workspace bump allocator
    char* ws = (char*)d_ws;
    size_t off = 0;
    auto alloc = [&](size_t bytes) -> void* {
        void* p = ws + off;
        off = (off + bytes + 255) & ~(size_t)255;
        return p;
    };

    __bf16* xp    = (__bf16*)alloc((size_t)LB * LT * LD * 2);  // packed A slabs
    __bf16* wpih0 = (__bf16*)alloc((size_t)LH * LG * 2);
    __bf16* wphh0 = (__bf16*)alloc((size_t)LH * LG * 2);
    __bf16* wpih1 = (__bf16*)alloc((size_t)LH * LG * 2);
    __bf16* wphh1 = (__bf16*)alloc((size_t)LH * LG * 2);
    __bf16* gin   = (__bf16*)alloc((size_t)LT * LB * LG * 2);  // reused L0/L1
    __bf16* hall  = (__bf16*)alloc((size_t)LT * LB * LH * 2);  // packed h0 slabs
    __bf16* zslab = (__bf16*)alloc((size_t)LB * LH * 2);       // zeros
    __bf16* h1a   = (__bf16*)alloc((size_t)LB * LH * 2);       // packed ping
    __bf16* h1b   = (__bf16*)alloc((size_t)LB * LH * 2);       // packed pong
    float*  c0    = (float*)alloc((size_t)LB * LH * 4);
    float*  c1    = (float*)alloc((size_t)LB * LH * 4);
    float*  gates = (float*)alloc((size_t)LB * LG * 4);

    // ---- init state (must be fresh every call)
    {
        int nz = (LB * LH * 2) / 4;
        zero_u32_kernel<<<(nz + 255) / 256, 256, 0, stream>>>((unsigned*)zslab, nz);
        int nc = LB * LH;
        zero_u32_kernel<<<(nc + 255) / 256, 256, 0, stream>>>((unsigned*)c0, nc);
        zero_u32_kernel<<<(nc + 255) / 256, 256, 0, stream>>>((unsigned*)c1, nc);
    }

    // ---- phase 0: packing
    {
        int n = LB * LT * LD;
        pack_x_kernel<<<n / 256, 256, 0, stream>>>(x, xp);
        int nw = LG * LH;
        pack_weight_kernel<<<nw / 256, 256, 0, stream>>>(w_ih_0, wpih0);
        pack_weight_kernel<<<nw / 256, 256, 0, stream>>>(w_hh_0, wphh0);
        pack_weight_kernel<<<nw / 256, 256, 0, stream>>>(w_ih_1, wpih1);
        pack_weight_kernel<<<nw / 256, 256, 0, stream>>>(w_hh_1, wphh1);
    }

    const dim3 bigGrid(LG / 128, LT);        // slab per t
    const dim3 stepGrid(LG / 64, 1);

    // ---- phase 1: Gin = x @ w_ih0^T
    gemm_wmma_packed<128, 8><<<bigGrid, 256, 0, stream>>>(
        xp, wpih0, gin, nullptr, nullptr);

    // ---- phase 2: layer-0 recurrence
    for (int t = 0; t < LT; ++t) {
        const __bf16* hprev =
            (t == 0) ? zslab : (hall + (size_t)(t - 1) * LB * LH);
        gemm_wmma_packed<64, 4><<<stepGrid, 128, 0, stream>>>(
            hprev, wphh0, nullptr, gates, gin + (size_t)t * LB * LG);
        bool last = (t == LT - 1);
        lstm_pointwise_kernel<<<(LB * LH) / 256, 256, 0, stream>>>(
            gates, c0, hall + (size_t)t * LB * LH, nullptr, t,
            last ? out + OFF_H0 : nullptr, last ? out + OFF_C0 : nullptr);
    }

    // ---- phase 3: Gin = h0_all @ w_ih1^T (hall already packed slabs)
    gemm_wmma_packed<128, 8><<<bigGrid, 256, 0, stream>>>(
        hall, wpih1, gin, nullptr, nullptr);

    // ---- phase 4: layer-1 recurrence (writes top-layer h to out)
    for (int t = 0; t < LT; ++t) {
        const __bf16* hprev = (t == 0) ? zslab : ((t & 1) ? h1a : h1b);
        __bf16* hnew = (t & 1) ? h1b : h1a;
        gemm_wmma_packed<64, 4><<<stepGrid, 128, 0, stream>>>(
            hprev, wphh1, nullptr, gates, gin + (size_t)t * LB * LG);
        bool last = (t == LT - 1);
        lstm_pointwise_kernel<<<(LB * LH) / 256, 256, 0, stream>>>(
            gates, c1, hnew, out + OUT_BTH, t,
            last ? out + OFF_H1 : nullptr, last ? out + OFF_C1 : nullptr);
    }
}
